// BaseLoftqLinear_18683107738177
// MI455X (gfx1250) — matrix operand order repaired and checked by reading, not compiled, not run on hardware
//
#include <hip/hip_runtime.h>
#include <stdint.h>

// ---------------------------------------------------------------------------
// Problem constants (from reference)
// ---------------------------------------------------------------------------
#define IN_F   4096
#define OUT_F  4096
#define M_TOT  8192           // 4 * 2048
#define RANK   16
#define QBLOCK 64

typedef __bf16 bf16;
typedef __attribute__((ext_vector_type(16))) __bf16 v16bf;
typedef __attribute__((ext_vector_type(8)))  float  v8f;

// ---------------------------------------------------------------------------
// CDNA5 async global->LDS DMA (ASYNCcnt-tracked), via inline asm.
// VFLAT form: global_load_async_to_lds_b128 vdst(LDS byte addr), vaddr64, off
// Generic->LDS address: hardware uses addr[31:0] as the LDS byte address.
// ---------------------------------------------------------------------------
__device__ __forceinline__ void async_copy_b128(const void* gsrc, uint32_t lds_addr) {
    asm volatile("global_load_async_to_lds_b128 %0, %1, off"
                 :
                 : "v"(lds_addr), "v"((uint64_t)(uintptr_t)gsrc)
                 : "memory");
}
__device__ __forceinline__ void wait_asynccnt0() {
    asm volatile("s_wait_asynccnt 0x0" ::: "memory");
}

// ---------------------------------------------------------------------------
// Kernel 1: dequantize 4-bit weights, fold LoRA (W_eff = W + B@A), store bf16
// Each thread handles 8 consecutive weights (= 4 packed bytes, one uint32).
// 8-aligned chunks never cross a 64-element scale block.
// ---------------------------------------------------------------------------
__global__ __launch_bounds__(256)
void dequant_fold_kernel(const uint8_t* __restrict__ qw,
                         const float*   __restrict__ wmax,
                         const float*   __restrict__ lut,
                         const float*   __restrict__ lA,   // [RANK, IN_F]
                         const float*   __restrict__ lB,   // [OUT_F, RANK]
                         bf16*          __restrict__ Wb) { // [OUT_F, IN_F]
    __shared__ float slut[16];
    if (threadIdx.x < 16) slut[threadIdx.x] = lut[threadIdx.x];
    __syncthreads();

    const int g  = blockIdx.x * 256 + threadIdx.x;   // group of 8 weights
    const int w0 = g * 8;
    const int o  = w0 >> 12;                          // / IN_F
    const int i0 = w0 & (IN_F - 1);

    const uint32_t pk    = ((const uint32_t*)qw)[g];
    const float    scale = wmax[w0 >> 6];

    float acc[8];
#pragma unroll
    for (int j = 0; j < 8; ++j)
        acc[j] = slut[(pk >> (4 * j)) & 15] * scale;

    // rank-16 LoRA fold
    const float* Bp = lB + o * RANK;
#pragma unroll
    for (int r = 0; r < RANK; ++r) {
        const float  bv = Bp[r];
        const float4 a0 = *(const float4*)(lA + (size_t)r * IN_F + i0);
        const float4 a1 = *(const float4*)(lA + (size_t)r * IN_F + i0 + 4);
        acc[0] += bv * a0.x; acc[1] += bv * a0.y;
        acc[2] += bv * a0.z; acc[3] += bv * a0.w;
        acc[4] += bv * a1.x; acc[5] += bv * a1.y;
        acc[6] += bv * a1.z; acc[7] += bv * a1.w;
    }

    union { uint4 u; bf16 h[8]; } pkd;
#pragma unroll
    for (int j = 0; j < 8; ++j) pkd.h[j] = (bf16)acc[j];
    *(uint4*)(Wb + w0) = pkd.u;
}

// ---------------------------------------------------------------------------
// Kernel 2: fp32 -> bf16 conversion of activations
// ---------------------------------------------------------------------------
__global__ __launch_bounds__(256)
void xcvt_kernel(const float* __restrict__ x, bf16* __restrict__ Xb) {
    const size_t i = ((size_t)blockIdx.x * 256 + threadIdx.x) * 8;
    const float4 f0 = *(const float4*)(x + i);
    const float4 f1 = *(const float4*)(x + i + 4);
    union { uint4 u; bf16 h[8]; } p;
    p.h[0] = (bf16)f0.x; p.h[1] = (bf16)f0.y;
    p.h[2] = (bf16)f0.z; p.h[3] = (bf16)f0.w;
    p.h[4] = (bf16)f1.x; p.h[5] = (bf16)f1.y;
    p.h[6] = (bf16)f1.z; p.h[7] = (bf16)f1.w;
    *(uint4*)(Xb + i) = p.u;
}

// ---------------------------------------------------------------------------
// Kernel 3: bf16 WMMA GEMM   out[m,n] = sum_k Xb[m,k]*Wb[n,k] + bias[n]
// 256 threads = 8 wave32s, tile 128x128, BK=32, double-buffered LDS.
// Tile staging: GLOBAL_LOAD_ASYNC_TO_LDS_B128 (ASYNCcnt DMA, no VGPR staging).
// ---------------------------------------------------------------------------
#define BM 128
#define BN 128
#define BK 32
#define LDS_STRIDE 80   // 64 data bytes per row + 16 pad (conflict-free b128)

union FragB16 { uint4 u[2]; v16bf v; };

// A fragment (16x32 bf16): lane l<16 -> row l, K {0..7,16..23}; l>=16 -> K {8..15,24..31}
__device__ inline v16bf load_frag_a(const uint8_t* base, int lane) {
    const uint8_t* p = base + (lane & 15) * LDS_STRIDE + (lane >> 4) * 16;
    FragB16 f;
    f.u[0] = *(const uint4*)(p);
    f.u[1] = *(const uint4*)(p + 32);
    return f.v;
}
// B fragment (32x16 bf16, from row-major W): lane n<16 -> W row n, K {0..7,16..23}; n>=16 -> K {8..15,24..31}
__device__ inline v16bf load_frag_b(const uint8_t* base, int lane) {
    const uint8_t* p = base + (lane & 15) * LDS_STRIDE + (lane >> 4) * 16;
    FragB16 f;
    f.u[0] = *(const uint4*)(p);
    f.u[1] = *(const uint4*)(p + 32);
    return f.v;
}

__global__ __launch_bounds__(256)
void gemm_bf16_wmma_kernel(const bf16* __restrict__ Xb,
                           const bf16* __restrict__ Wb,
                           const float* __restrict__ bias,
                           float* __restrict__ out) {
    __shared__ __attribute__((aligned(16))) uint8_t lds[2][2][BM * LDS_STRIDE];

    const int m0   = blockIdx.y * BM;
    const int n0   = blockIdx.x * BN;
    const int t    = threadIdx.x;
    const int lane = t & 31;
    const int wid  = t >> 5;
    const int wm   = wid >> 2;      // 0..1 -> row offset wm*64
    const int wn   = wid & 3;       // 0..3 -> col offset wn*32

    // global tile loader mapping: 512 chunks of 16B per matrix, 2 per thread
    const int lr = t >> 2;          // rows lr and lr+64
    const int lc = (t & 3) * 16;    // byte column within 64B row segment

    const size_t rowBytes = (size_t)IN_F * 2;
    const uint8_t* gA = (const uint8_t*)Xb + (size_t)m0 * rowBytes;
    const uint8_t* gB = (const uint8_t*)Wb + (size_t)n0 * rowBytes;
    const uint8_t* gA0 = gA + (size_t)lr        * rowBytes + lc;
    const uint8_t* gA1 = gA + (size_t)(lr + 64) * rowBytes + lc;
    const uint8_t* gB0 = gB + (size_t)lr        * rowBytes + lc;
    const uint8_t* gB1 = gB + (size_t)(lr + 64) * rowBytes + lc;

    // per-thread LDS destinations (byte addresses, low 32 bits of generic ptr)
    const uint32_t ldsA0[2] = {
        (uint32_t)(uintptr_t)&lds[0][0][lr * LDS_STRIDE + lc],
        (uint32_t)(uintptr_t)&lds[1][0][lr * LDS_STRIDE + lc] };
    const uint32_t ldsA1[2] = {
        (uint32_t)(uintptr_t)&lds[0][0][(lr + 64) * LDS_STRIDE + lc],
        (uint32_t)(uintptr_t)&lds[1][0][(lr + 64) * LDS_STRIDE + lc] };
    const uint32_t ldsB0[2] = {
        (uint32_t)(uintptr_t)&lds[0][1][lr * LDS_STRIDE + lc],
        (uint32_t)(uintptr_t)&lds[1][1][lr * LDS_STRIDE + lc] };
    const uint32_t ldsB1[2] = {
        (uint32_t)(uintptr_t)&lds[0][1][(lr + 64) * LDS_STRIDE + lc],
        (uint32_t)(uintptr_t)&lds[1][1][(lr + 64) * LDS_STRIDE + lc] };

    // prologue: DMA K-tile 0 into stage 0
    async_copy_b128(gA0, ldsA0[0]);
    async_copy_b128(gA1, ldsA1[0]);
    async_copy_b128(gB0, ldsB0[0]);
    async_copy_b128(gB1, ldsB1[0]);
    wait_asynccnt0();
    __syncthreads();

    v8f acc[4][2] = {};
    const int NK = IN_F / BK;       // 128 K-steps

    for (int kt = 0; kt < NK; ++kt) {
        const int cur = kt & 1;
        const int nxt = cur ^ 1;
        const bool have_next = (kt + 1 < NK);

        if (have_next) {
            // DMA next tile directly into the other LDS stage (no VGPRs)
            const size_t kb = (size_t)(kt + 1) * (BK * 2);
            async_copy_b128(gA0 + kb, ldsA0[nxt]);
            async_copy_b128(gA1 + kb, ldsA1[nxt]);
            async_copy_b128(gB0 + kb, ldsB0[nxt]);
            async_copy_b128(gB1 + kb, ldsB1[nxt]);
            if (kt + 4 < NK) {       // pull future tiles toward L2
                const size_t kp = (size_t)(kt + 4) * (BK * 2);
                __builtin_prefetch(gA0 + kp, 0, 1);
                __builtin_prefetch(gB0 + kp, 0, 1);
            }
        }

        // compute from current stage
        const uint8_t* lA = &lds[cur][0][0];
        const uint8_t* lB = &lds[cur][1][0];

        v16bf af[4], bfr[2];
#pragma unroll
        for (int i = 0; i < 4; ++i)
            af[i] = load_frag_a(lA + (wm * 64 + i * 16) * LDS_STRIDE, lane);
#pragma unroll
        for (int j = 0; j < 2; ++j)
            bfr[j] = load_frag_b(lB + (wn * 32 + j * 16) * LDS_STRIDE, lane);

#pragma unroll
        for (int i = 0; i < 4; ++i)
#pragma unroll
            for (int j = 0; j < 2; ++j)
                acc[i][j] = __builtin_amdgcn_wmma_f32_16x16x32_bf16(
                    false, af[i], false, bfr[j], (short)0, acc[i][j],
                    false, false);

        if (have_next) wait_asynccnt0();
        __syncthreads();
    }

    // epilogue: C layout — lane&15 = N, VGPR v = M%8, lane>>4 selects M+8
    const int hl = lane & 15;
    const int hh = lane >> 4;
#pragma unroll
    for (int i = 0; i < 4; ++i) {
#pragma unroll
        for (int j = 0; j < 2; ++j) {
            const int col   = n0 + wn * 32 + j * 16 + hl;
            const float bv  = bias[col];
            const int rbase = m0 + wm * 64 + i * 16 + hh * 8;
#pragma unroll
            for (int v = 0; v < 8; ++v)
                out[(size_t)(rbase + v) * OUT_F + col] = acc[i][j][v] + bv;
        }
    }
}

// ---------------------------------------------------------------------------
// Launch
// ---------------------------------------------------------------------------
extern "C" void kernel_launch(void* const* d_in, const int* in_sizes, int n_in,
                              void* d_out, int out_size, void* d_ws, size_t ws_size,
                              hipStream_t stream) {
    const float*   x    = (const float*)d_in[0];
    const uint8_t* qw   = (const uint8_t*)d_in[1];
    const float*   wmax = (const float*)d_in[2];
    const float*   lut  = (const float*)d_in[3];
    const float*   lA   = (const float*)d_in[4];
    const float*   lB   = (const float*)d_in[5];
    const float*   bias = (const float*)d_in[6];
    float*         out  = (float*)d_out;

    // workspace layout: [W_eff bf16: 32 MB][x bf16: 64 MB]
    bf16* Wb = (bf16*)d_ws;
    bf16* Xb = (bf16*)((char*)d_ws + (size_t)OUT_F * IN_F * sizeof(bf16));

    dequant_fold_kernel<<<(OUT_F * IN_F / 8) / 256, 256, 0, stream>>>(
        qw, wmax, lut, lA, lB, Wb);

    xcvt_kernel<<<((size_t)M_TOT * IN_F / 8) / 256, 256, 0, stream>>>(x, Xb);

    dim3 grid(OUT_F / BN, M_TOT / BM);  // 32 x 64 = 2048 workgroups
    gemm_bf16_wmma_kernel<<<grid, 256, 0, stream>>>(Xb, Wb, bias, out);
}
